// DogeDecoderLayer_85787676770822
// MI455X (gfx1250) — matrix operand order repaired and checked by reading, not compiled, not run on hardware
//
#include <hip/hip_runtime.h>
#include <hip/hip_bf16.h>
#include <math.h>

#define B_ 2
#define S_ 1024
#define DM_ 2048
#define H_ 16
#define HK_ 8
#define DH_ 128
#define KV_ 1024
#define FF_ 8192
#define NROWS_ (B_*S_)          // 2048
#define SCALE_ 0.08838834764831845f
#define MINVAL_ -3.4028235e38f

typedef __attribute__((ext_vector_type(16))) __bf16 v16bf;
typedef __attribute__((ext_vector_type(8)))  float  v8f;

union Frag { uint4 q[2]; v16bf v; };

__device__ __forceinline__ unsigned short f2bf(float f) {
  union { float f; unsigned u; } x; x.f = f;
  return (unsigned short)((x.u + 0x7fffu + ((x.u >> 16) & 1u)) >> 16);
}
__device__ __forceinline__ float bf2f(unsigned short h) {
  union { unsigned u; float f; } x; x.u = ((unsigned)h) << 16; return x.f;
}
__device__ __forceinline__ v8f wmma_bf16(v16bf a, v16bf b, v8f c) {
  return __builtin_amdgcn_wmma_f32_16x16x32_bf16(false, a, false, b, (short)0, c, false, false);
}

// Async copy of 16B global -> LDS (gfx1250, ASYNCcnt-tracked).
__device__ __forceinline__ void async_g2l_b128(unsigned lds_byte_off, const void* gptr) {
  asm volatile("global_load_async_to_lds_b128 %0, %1, off"
               :: "v"(lds_byte_off), "v"((unsigned long long)(uintptr_t)gptr)
               : "memory");
}

// ---------------------------------------------------------------------------
// Pack an fp32 [K x N] weight into bf16 B-fragment layout:
// per 32(K)x16(N) tile, lane = nn + 16*(kk>>4), half = kk&15 (16 contiguous
// halves per lane -> two contiguous b128 loads in the GEMM).
// ---------------------------------------------------------------------------
__global__ void packw_kernel(const float* __restrict__ W,
                             unsigned short* __restrict__ Wp, int K, int N) {
  size_t i = (size_t)blockIdx.x * 256 + threadIdx.x;
  if (i >= (size_t)K * N) return;
  int n = (int)(i % N), k = (int)(i / N);
  int kb = k >> 5, kk = k & 31, nb = n >> 4, nn = n & 15;
  int lane = nn + ((kk >> 4) << 4);
  int h    = kk & 15;
  Wp[(((size_t)kb * (N >> 4) + nb) * 32 + lane) * 16 + h] = f2bf(W[i]);
}

// ---------------------------------------------------------------------------
// RMSNorm (fp32 in) -> bf16 out. One block per row.
// ---------------------------------------------------------------------------
__global__ __launch_bounds__(256) void rmsnorm_bf16_kernel(
    const float* __restrict__ X, const float* __restrict__ W,
    unsigned short* __restrict__ Y, int D) {
  __shared__ float red[8];
  __shared__ float stot;
  int row = blockIdx.x;
  const float* x = X + (size_t)row * D;
  float ss = 0.f;
  for (int i = threadIdx.x; i < D; i += 256) { float v = x[i]; ss += v * v; }
  for (int m = 16; m >= 1; m >>= 1) ss += __shfl_xor(ss, m, 32);
  if ((threadIdx.x & 31) == 0) red[threadIdx.x >> 5] = ss;
  __syncthreads();
  if (threadIdx.x == 0) { float t = 0.f; for (int i = 0; i < 8; i++) t += red[i]; stot = t; }
  __syncthreads();
  float rstd = rsqrtf(stot / (float)D + 1e-6f);
  for (int i = threadIdx.x; i < D; i += 256)
    Y[(size_t)row * D + i] = f2bf(x[i] * rstd * W[i]);
}

// ---------------------------------------------------------------------------
// bf16 WMMA GEMM: C[M,N] = A[M,K] @ B[K,N].
// Block 256 thr = 8 waves; block tile 128x128; wave tile 32x64 (2x4 frags).
// A tile (128x32 bf16 = 8KB) is double-buffered in LDS and filled with
// global_load_async_to_lds_b128 (ASYNCcnt) so the async DMA overlaps WMMAs.
// B is fragment-packed bf16, read register-direct (L2-resident).
// mode 0: store fp32   mode 1: store fp32 += resw[col]*res[row,col]
// mode 2: store bf16.
// ---------------------------------------------------------------------------
__global__ __launch_bounds__(256) void gemm_bf16_wmma(
    const unsigned short* __restrict__ A, const unsigned short* __restrict__ Bp,
    float* __restrict__ Cf, unsigned short* __restrict__ Cb,
    const float* __restrict__ res, const float* __restrict__ resw,
    int N, int K, int mode) {
  __shared__ __align__(16) unsigned short Atile[2][128 * 32];
  const int lane = threadIdx.x & 31;
  const int w = threadIdx.x >> 5;
  const int mblk = blockIdx.y * 128;
  const int m0w = (w & 3) * 32;                 // wave row offset in block tile
  const int n0 = blockIdx.x * 128 + (w >> 2) * 64;
  const int r = lane & 15;
  const int kbA = (lane >> 4) << 4;             // A-layout K base per lane-half
  const int nT = N >> 4;
  const unsigned ldsbase = (unsigned)(uintptr_t)(&Atile[0][0]);

  // per-thread async-fill assignment: 2 x b128 per thread per tile
  const int e0 = threadIdx.x;                   // 0..255
  const int row0 = e0 >> 2, seg0 = e0 & 3;
  const int e1 = threadIdx.x + 256;             // 256..511
  const int row1 = e1 >> 2, seg1 = e1 & 3;

  auto issue_tile = [&](int buf, int k0) {
    async_g2l_b128(ldsbase + (unsigned)buf * 8192u + (unsigned)(row0 * 64 + seg0 * 16),
                   A + (size_t)(mblk + row0) * K + k0 + seg0 * 8);
    async_g2l_b128(ldsbase + (unsigned)buf * 8192u + (unsigned)(row1 * 64 + seg1 * 16),
                   A + (size_t)(mblk + row1) * K + k0 + seg1 * 8);
  };

  v8f acc[2][4];
  #pragma unroll
  for (int i = 0; i < 2; i++)
    #pragma unroll
    for (int j = 0; j < 4; j++)
      #pragma unroll
      for (int e = 0; e < 8; e++) acc[i][j][e] = 0.f;

  const int T = K >> 5;                         // number of 32-wide K tiles
  issue_tile(0, 0);

  for (int t = 0; t < T; t++) {
    const int k0 = t << 5;
    if (t + 1 < T) {
      issue_tile((t + 1) & 1, k0 + 32);
      asm volatile("s_wait_asynccnt 0x2" ::: "memory");
    } else {
      asm volatile("s_wait_asynccnt 0x0" ::: "memory");
    }
    __syncthreads();                            // tile t visible to all waves

    const unsigned short* As = &Atile[t & 1][0];
    Frag a[2], b[4];
    #pragma unroll
    for (int i = 0; i < 2; i++) {
      const unsigned short* p = As + (m0w + i * 16 + r) * 32 + kbA;
      a[i].q[0] = *(const uint4*)p;
      a[i].q[1] = *(const uint4*)(p + 8);
    }
    #pragma unroll
    for (int j = 0; j < 4; j++) {
      const unsigned short* p =
          Bp + (((size_t)t * nT + ((n0 >> 4) + j)) * 32 + lane) * 16;
      b[j].q[0] = *(const uint4*)p;
      b[j].q[1] = *(const uint4*)(p + 8);
    }
    if (t + 1 < T)
      __builtin_prefetch(Bp + (((size_t)(t + 1) * nT + (n0 >> 4)) * 32 + lane) * 16, 0, 1);
    #pragma unroll
    for (int i = 0; i < 2; i++)
      #pragma unroll
      for (int j = 0; j < 4; j++)
        acc[i][j] = wmma_bf16(a[i].v, b[j].v, acc[i][j]);

    __syncthreads();                            // done reading tile t before refill
  }

  const int rowHalf = (lane >> 4) << 3;
  #pragma unroll
  for (int i = 0; i < 2; i++)
    #pragma unroll
    for (int j = 0; j < 4; j++) {
      int col = n0 + j * 16 + r;
      #pragma unroll
      for (int v = 0; v < 8; v++) {
        int row = mblk + m0w + i * 16 + rowHalf + v;
        float val = acc[i][j][v];
        size_t idx = (size_t)row * N + col;
        if (mode == 1) val = fmaf(resw[col], res[idx], val);
        if (mode == 2) Cb[idx] = f2bf(val);
        else           Cf[idx] = val;
      }
    }
}

// ---------------------------------------------------------------------------
// Per-head RMSNorm + RoPE: fp32 [rows, nh*DH] -> bf16 same layout.
// One block (128 thr) per (row, head). Position = s (= row % S).
// ---------------------------------------------------------------------------
__global__ __launch_bounds__(128) void qknorm_rope_kernel(
    const float* __restrict__ X, const float* __restrict__ W,
    unsigned short* __restrict__ Y, int nh) {
  __shared__ float red[4];
  int idx = blockIdx.x;
  int head = idx % nh, row = idx / nh;
  int s = row & (S_ - 1);
  const float* x = X + ((size_t)row * nh + head) * DH_;
  int d = threadIdx.x;
  float v = x[d];
  float ss = v * v;
  for (int m = 16; m >= 1; m >>= 1) ss += __shfl_xor(ss, m, 32);
  if ((threadIdx.x & 31) == 0) red[threadIdx.x >> 5] = ss;
  __syncthreads();
  float tot = red[0] + red[1] + red[2] + red[3];
  float rstd = rsqrtf(tot / 128.f + 1e-6f);
  float xn = v * rstd * W[d];
  int d2 = (d < 64) ? d + 64 : d - 64;
  float v2 = x[d2] * rstd * W[d2];
  float rot = (d < 64) ? -v2 : v2;
  int i = d & 63;
  float freq = (float)s * __powf(10000.f, -(float)(2 * i) / 128.f);
  float sn, cs;
  __sincosf(freq, &sn, &cs);
  Y[((size_t)row * nh + head) * DH_ + d] = f2bf(xn * cs + rot * sn);
}

// ---------------------------------------------------------------------------
// V transpose to dh-major bf16: fp32 [B,S,HK,DH] -> bf16 [B,HK,DH,S]
// (so attention P@V B-fragment loads are contiguous over keys).
// ---------------------------------------------------------------------------
__global__ void vpack_kernel(const float* __restrict__ V,
                             unsigned short* __restrict__ Vt) {
  size_t i = (size_t)blockIdx.x * 256 + threadIdx.x;   // over B*S*HK*DH
  int d = (int)(i & 127);
  size_t t = i >> 7;
  int hk = (int)(t & 7);
  size_t bs = t >> 3;
  int s = (int)(bs & (S_ - 1));
  int b = (int)(bs >> 10);
  Vt[(((size_t)b * HK_ + hk) * DH_ + d) * S_ + s] = f2bf(V[i]);
}

// ---------------------------------------------------------------------------
// dt bias: dt[b,hk,s] = exp(A[hk] * softplus(v_flat[b,s,:] . dt_w[:,hk]))
// One block (256 thr = 8 hk * 32 lanes) per (b,s).
// ---------------------------------------------------------------------------
__global__ __launch_bounds__(256) void dt_kernel(
    const float* __restrict__ Vf, const float* __restrict__ dtw,
    const float* __restrict__ A, float* __restrict__ dtb) {
  int row = blockIdx.x;                      // b*S+s
  int hk = threadIdx.x >> 5, l = threadIdx.x & 31;
  const float* v = Vf + (size_t)row * KV_;
  float acc = 0.f;
  for (int i = l; i < KV_; i += 32) acc = fmaf(v[i], dtw[i * HK_ + hk], acc);
  for (int m = 16; m >= 1; m >>= 1) acc += __shfl_xor(acc, m, 32);
  if (l == 0) {
    float sp = (acc > 20.f) ? acc : log1pf(__expf(acc));
    int b = row >> 10, s = row & (S_ - 1);
    dtb[((size_t)b * HK_ + hk) * S_ + s] = __expf(A[hk] * sp);
  }
}

// ---------------------------------------------------------------------------
// Flash attention with WMMA. One wave per (b, h, 16-query tile).
// Q bf16 [B,S,H,DH], K bf16 [B,S,HK,DH], Vt bf16 [B,HK,DH,S], dtb [B,HK,S].
// Output bf16 [B,S,H*DH].
// ---------------------------------------------------------------------------
__global__ __launch_bounds__(128) void attn_kernel(
    const unsigned short* __restrict__ Qb, const unsigned short* __restrict__ Kb,
    const unsigned short* __restrict__ Vt, const float* __restrict__ dtb,
    unsigned short* __restrict__ Ob) {
  __shared__ __align__(16) unsigned short lp[4][16 * 32];
  int lane = threadIdx.x & 31;
  int w = threadIdx.x >> 5;
  int gw = blockIdx.x * 4 + w;          // 0 .. B*H*(S/16)-1
  int qt = gw & 63;
  int h  = (gw >> 6) & 15;
  int b  = gw >> 10;
  int hk = h >> 1;
  int r = lane & 15;
  int kbA = (lane >> 4) << 4;           // A-frag K base
  int kbB = (lane >> 4) << 4;           // B-frag K base

  Frag qf[4];
  #pragma unroll
  for (int c = 0; c < 4; c++) {
    const unsigned short* p =
        Qb + (((size_t)(b * S_ + qt * 16 + r) * H_) + h) * DH_ + c * 32 + kbA;
    qf[c].q[0] = *(const uint4*)p;
    qf[c].q[1] = *(const uint4*)(p + 8);
  }

  float rmax[8], rsum[8];
  #pragma unroll
  for (int i = 0; i < 8; i++) { rmax[i] = MINVAL_; rsum[i] = 0.f; }
  v8f oacc[8];
  #pragma unroll
  for (int t = 0; t < 8; t++)
    #pragma unroll
    for (int e = 0; e < 8; e++) oacc[t][e] = 0.f;

  int kend = qt * 16 + 16;
  for (int j0 = 0; j0 < kend; j0 += 32) {
    float sc[2][8];
    #pragma unroll
    for (int jj = 0; jj < 2; jj++) {
      v8f s;
      #pragma unroll
      for (int e = 0; e < 8; e++) s[e] = 0.f;
      int key = j0 + jj * 16 + r;       // this lane's key column
      #pragma unroll
      for (int c = 0; c < 4; c++) {
        Frag kf;
        const unsigned short* p =
            Kb + (((size_t)(b * S_ + key) * HK_) + hk) * DH_ + c * 32 + kbB;
        kf.q[0] = *(const uint4*)p;
        kf.q[1] = *(const uint4*)(p + 8);
        s = wmma_bf16(qf[c].v, kf.v, s);
      }
      float dtv = dtb[((size_t)b * HK_ + hk) * S_ + key];
      #pragma unroll
      for (int i = 0; i < 8; i++) {
        int qi = qt * 16 + i + ((lane >> 4) << 3);
        sc[jj][i] = (key <= qi) ? fmaf(s[i], SCALE_, dtv) : MINVAL_;
      }
    }
    // online softmax over the 32 new keys (reduce across 16 lanes per half)
    #pragma unroll
    for (int i = 0; i < 8; i++) {
      float t = fmaxf(sc[0][i], sc[1][i]);
      for (int m = 8; m >= 1; m >>= 1) t = fmaxf(t, __shfl_xor(t, m, 32));
      float mn = fmaxf(rmax[i], t);
      float corr = __expf(rmax[i] - mn);
      float p0 = __expf(sc[0][i] - mn);
      float p1 = __expf(sc[1][i] - mn);
      float ps = p0 + p1;
      for (int m = 8; m >= 1; m >>= 1) ps += __shfl_xor(ps, m, 32);
      rsum[i] = rsum[i] * corr + ps;
      rmax[i] = mn;
      #pragma unroll
      for (int t2 = 0; t2 < 8; t2++) oacc[t2][i] = oacc[t2][i] * corr;
      int row = i + ((lane >> 4) << 3);
      lp[w][row * 32 + r]      = f2bf(p0);
      lp[w][row * 32 + 16 + r] = f2bf(p1);
    }
    // reload P as an A-fragment (16 queries x 32 keys) and do P @ V
    Frag pf;
    const unsigned short* pp = &lp[w][r * 32 + kbA];
    pf.q[0] = *(const uint4*)pp;
    pf.q[1] = *(const uint4*)(pp + 8);
    #pragma unroll
    for (int t = 0; t < 8; t++) {
      Frag vf;
      const unsigned short* p =
          Vt + (((size_t)b * HK_ + hk) * DH_ + t * 16 + r) * S_ + j0 + kbB;
      vf.q[0] = *(const uint4*)p;
      vf.q[1] = *(const uint4*)(p + 8);
      oacc[t] = wmma_bf16(pf.v, vf.v, oacc[t]);
    }
  }

  #pragma unroll
  for (int i = 0; i < 8; i++) {
    float inv = 1.f / rsum[i];
    int row = qt * 16 + i + ((lane >> 4) << 3);
    #pragma unroll
    for (int t = 0; t < 8; t++)
      Ob[((size_t)(b * S_ + row)) * (H_ * DH_) + h * DH_ + t * 16 + r] =
          f2bf(oacc[t][i] * inv);
  }
}

// ---------------------------------------------------------------------------
// SiLU(gate) * up, bf16 in/out.
// ---------------------------------------------------------------------------
__global__ void silumul_kernel(const unsigned short* __restrict__ g,
                               const unsigned short* __restrict__ u,
                               unsigned short* __restrict__ o, size_t n) {
  size_t i = (size_t)blockIdx.x * 256 + threadIdx.x;
  if (i >= n) return;
  float gf = bf2f(g[i]), uf = bf2f(u[i]);
  float s = gf / (1.f + __expf(-gf));
  o[i] = f2bf(s * uf);
}

// ---------------------------------------------------------------------------
extern "C" void kernel_launch(void* const* d_in, const int* in_sizes, int n_in,
                              void* d_out, int out_size, void* d_ws, size_t ws_size,
                              hipStream_t stream) {
  (void)in_sizes; (void)n_in; (void)out_size; (void)ws_size;
  const float* hidden   = (const float*)d_in[0];
  const float* in_ln_w  = (const float*)d_in[1];
  const float* q_w      = (const float*)d_in[2];
  const float* k_w      = (const float*)d_in[3];
  const float* v_w      = (const float*)d_in[4];
  const float* o_w      = (const float*)d_in[5];
  const float* q_norm_w = (const float*)d_in[6];
  const float* k_norm_w = (const float*)d_in[7];
  const float* Avec     = (const float*)d_in[8];
  const float* dt_w     = (const float*)d_in[9];
  const float* in_res   = (const float*)d_in[10];
  const float* post_ln_w= (const float*)d_in[11];
  const float* gate_w   = (const float*)d_in[12];
  const float* up_w     = (const float*)d_in[13];
  const float* down_w   = (const float*)d_in[14];
  const float* post_res = (const float*)d_in[15];
  float* out = (float*)d_out;

  char* ws = (char*)d_ws;
  size_t off = 0;
  auto alloc = [&](size_t bytes) -> void* {
    void* p = ws + off;
    off += (bytes + 255) & ~(size_t)255;
    return p;
  };
  unsigned short* WqP = (unsigned short*)alloc((size_t)DM_*DM_*2);
  unsigned short* WkP = (unsigned short*)alloc((size_t)DM_*KV_*2);
  unsigned short* WvP = (unsigned short*)alloc((size_t)DM_*KV_*2);
  unsigned short* WoP = (unsigned short*)alloc((size_t)DM_*DM_*2);
  unsigned short* WgP = (unsigned short*)alloc((size_t)DM_*FF_*2);
  unsigned short* WuP = (unsigned short*)alloc((size_t)DM_*FF_*2);
  unsigned short* WdP = (unsigned short*)alloc((size_t)FF_*DM_*2);
  unsigned short* Xbf = (unsigned short*)alloc((size_t)NROWS_*DM_*2);
  float*          Qf  = (float*)alloc((size_t)NROWS_*DM_*4);
  float*          Kf  = (float*)alloc((size_t)NROWS_*KV_*4);
  float*          Vf  = (float*)alloc((size_t)NROWS_*KV_*4);
  unsigned short* Qbf = (unsigned short*)alloc((size_t)NROWS_*DM_*2);
  unsigned short* Kbf = (unsigned short*)alloc((size_t)NROWS_*KV_*2);
  unsigned short* VtT = (unsigned short*)alloc((size_t)NROWS_*KV_*2);
  float*          Dtb = (float*)alloc((size_t)B_*HK_*S_*4);
  unsigned short* Atn = (unsigned short*)alloc((size_t)NROWS_*DM_*2);
  float*          Hf  = (float*)alloc((size_t)NROWS_*DM_*4);
  unsigned short* Hnb = (unsigned short*)alloc((size_t)NROWS_*DM_*2);
  unsigned short* Gbf = (unsigned short*)alloc((size_t)NROWS_*FF_*2);
  unsigned short* Ubf = (unsigned short*)alloc((size_t)NROWS_*FF_*2);
  unsigned short* Mbf = (unsigned short*)alloc((size_t)NROWS_*FF_*2);

  auto packw = [&](const float* src, unsigned short* dst, int K, int N) {
    size_t n = (size_t)K * N;
    packw_kernel<<<(unsigned)((n + 255) / 256), 256, 0, stream>>>(src, dst, K, N);
  };
  auto gemm = [&](const unsigned short* A, const unsigned short* Bp,
                  float* Cf, unsigned short* Cb, const float* res,
                  const float* resw, int N, int K, int mode) {
    dim3 g(N / 128, NROWS_ / 128);
    gemm_bf16_wmma<<<g, 256, 0, stream>>>(A, Bp, Cf, Cb, res, resw, N, K, mode);
  };

  // 1) weights -> packed bf16 (120 MB total: fits MI455X's 192 MB L2)
  packw(q_w,   WqP, DM_, DM_);
  packw(k_w,   WkP, DM_, KV_);
  packw(v_w,   WvP, DM_, KV_);
  packw(o_w,   WoP, DM_, DM_);
  packw(gate_w,WgP, DM_, FF_);
  packw(up_w,  WuP, DM_, FF_);
  packw(down_w,WdP, FF_, DM_);

  // 2) input RMSNorm -> bf16 activations
  rmsnorm_bf16_kernel<<<NROWS_, 256, 0, stream>>>(hidden, in_ln_w, Xbf, DM_);

  // 3) Q/K/V projections (bf16 WMMA, f32 accumulate, async A staging)
  gemm(Xbf, WqP, Qf, nullptr, nullptr, nullptr, DM_, DM_, 0);
  gemm(Xbf, WkP, Kf, nullptr, nullptr, nullptr, KV_, DM_, 0);
  gemm(Xbf, WvP, Vf, nullptr, nullptr, nullptr, KV_, DM_, 0);

  // 4) per-head RMSNorm + RoPE -> bf16
  qknorm_rope_kernel<<<NROWS_ * H_, 128, 0, stream>>>(Qf, q_norm_w, Qbf, H_);
  qknorm_rope_kernel<<<NROWS_ * HK_, 128, 0, stream>>>(Kf, k_norm_w, Kbf, HK_);

  // 5) V -> dh-major bf16, dt bias
  vpack_kernel<<<(B_*S_*KV_) / 256, 256, 0, stream>>>(Vf, VtT);
  dt_kernel<<<NROWS_, 256, 0, stream>>>(Vf, dt_w, Avec, Dtb);

  // 6) flash attention (WMMA QK^T and P·V)
  attn_kernel<<<(B_ * H_ * (S_ / 16)) / 4, 128, 0, stream>>>(Qbf, Kbf, VtT, Dtb, Atn);

  // 7) O-proj + residual -> h (fp32)
  gemm(Atn, WoP, Hf, nullptr, hidden, in_res, DM_, DM_, 1);

  // 8) post RMSNorm -> bf16
  rmsnorm_bf16_kernel<<<NROWS_, 256, 0, stream>>>(Hf, post_ln_w, Hnb, DM_);

  // 9) gate/up GEMMs (bf16 out), SiLU-mul
  gemm(Hnb, WgP, nullptr, Gbf, nullptr, nullptr, FF_, DM_, 2);
  gemm(Hnb, WuP, nullptr, Ubf, nullptr, nullptr, FF_, DM_, 2);
  silumul_kernel<<<(unsigned)(((size_t)NROWS_*FF_ + 255) / 256), 256, 0, stream>>>(
      Gbf, Ubf, Mbf, (size_t)NROWS_ * FF_);

  // 10) down GEMM + residual -> final fp32 output
  gemm(Mbf, WdP, out, nullptr, Hf, post_res, DM_, FF_, 1);
}
